// LSTMDecoder_6528350290266
// MI455X (gfx1250) — matrix operand order repaired and checked by reading, not compiled, not run on hardware
//
#include <hip/hip_runtime.h>
#include <hip/hip_bf16.h>

// LSTM decoder: B=32, S=512, D=512, H=1024 (gate order i,f,g,o).
// Persistent kernel, one grid barrier per timestep.
//  - Weights converted to bf16 once; W (12MB bf16) stays L2-resident (192MB L2).
//  - Per step, the activations A = [x_t | h_{t-1}] (32x1536 bf16) are staged
//    into LDS once per WG by the Tensor Data Mover (tensor_load_to_lds,
//    TENSORcnt), with TDM pad_enable inserting 16B per 1KB so row-parallel
//    ds_load_b128 fragment reads avoid 16-way bank conflicts.
//  - Gates computed with v_wmma_f32_16x16x32_bf16, fp32 accumulation.

#define B_   32
#define S_   512
#define D_   512
#define H_   1024
#define G4H  4096

typedef __attribute__((ext_vector_type(16))) __bf16 v16bf;
typedef __attribute__((ext_vector_type(8)))  __bf16 v8bf;
typedef __attribute__((ext_vector_type(8)))  float  v8f;
typedef __attribute__((ext_vector_type(4)))  unsigned int u32x4;
typedef __attribute__((ext_vector_type(8)))  int i32x8;
typedef __attribute__((ext_vector_type(4)))  int i32x4;

union FragU { v16bf v; v8bf h[2]; };

__device__ __forceinline__ v16bf load_frag(const void* p0, const void* p1) {
    FragU u;
    u.h[0] = *(const v8bf*)p0;
    u.h[1] = *(const v8bf*)p1;
    return u.v;
}

// ---- TDM: 2D tile load Global -> LDS, bf16 elements, optional LDS padding ----
// Pad policy (when pad=true): pad_interval=256 dwords (1KB), pad_amount=4 dwords
// (16B) -> row strides stay 16B aligned and bank-shift consecutive rows.
__device__ __forceinline__ void tdm_load_2d(unsigned lds_byte_addr,
                                            const void* gptr,
                                            unsigned dim0, unsigned dim1,
                                            unsigned tile0, unsigned tile1,
                                            unsigned long long stride0_elems,
                                            bool pad) {
    unsigned long long ga = (unsigned long long)gptr;
    u32x4 g0;
    g0[0] = 1u;                                   // count=1 (valid), user mode
    g0[1] = lds_byte_addr;                        // D#.lds_addr
    g0[2] = (unsigned)(ga & 0xFFFFFFFFu);         // global_addr[31:0]
    g0[3] = (unsigned)((ga >> 32) & 0x01FFFFFFu)  // global_addr[56:32]
          | (2u << 30);                           // type=2 ("image")
    unsigned w0 = (1u << 16);                     // data_size=1 -> 2 bytes
    if (pad) w0 |= (1u << 20)                     // pad_enable
               |  (7u << 22)                      // pad_interval: 256 dwords
               |  (3u << 25);                     // pad_amount:   4 dwords
    i32x8 g1;
    g1[0] = (int)w0;                              // wg_mask=0 (not in cluster)
    g1[1] = (int)((dim0 & 0xFFFFu) << 16);        // [31:16]=tensor_dim0 lo
    g1[2] = (int)(((dim0 >> 16) & 0xFFFFu) | ((dim1 & 0xFFFFu) << 16));
    g1[3] = (int)(((dim1 >> 16) & 0xFFFFu) | ((tile0 & 0xFFFFu) << 16));
    g1[4] = (int)(tile1 & 0xFFFFu);               // tile_dim1; tile_dim2=0
    g1[5] = (int)(unsigned)(stride0_elems & 0xFFFFFFFFull);
    g1[6] = (int)(unsigned)((stride0_elems >> 32) & 0xFFFFull);
    g1[7] = 0;
    i32x4 z4 = {0, 0, 0, 0};                      // 2D: groups 2/3 unused
    i32x8 z8 = {0, 0, 0, 0, 0, 0, 0, 0};
    __builtin_amdgcn_tensor_load_to_lds(g0, g1, z4, z4, z8, 0);
}

__global__ void cvt_f32_to_bf16(const float* __restrict__ src,
                                __bf16* __restrict__ dst, int n) {
    int i = blockIdx.x * blockDim.x + threadIdx.x;
    if (i < n) dst[i] = (__bf16)src[i];
}

__global__ void prep_kernel(const float* __restrict__ b_ih,
                            const float* __restrict__ b_hh,
                            float* __restrict__ bsum,
                            const float* __restrict__ h0,
                            const float* __restrict__ c0,
                            __bf16* __restrict__ hbuf,   // [2][B][H], fill buf 0
                            float* __restrict__ c,       // [B][H]
                            unsigned* __restrict__ bar) {
    int i = blockIdx.x * blockDim.x + threadIdx.x;
    if (i < G4H) bsum[i] = b_ih[i] + b_hh[i];
    if (i < B_ * H_) {
        hbuf[i] = (__bf16)h0[i];
        c[i]    = c0[i];
    }
    if (i == 0) *bar = 0u;
}

// LDS staging layout (bytes). x rows: 512 bf16 = 1024B + 16B pad -> 1040.
// h rows: 1024 bf16 = 2048B + 2x16B pads (at 1KB boundaries) -> 2080.
#define XS_STRIDE 1040
#define HS_STRIDE 2080
#define XS_BYTES  (B_ * XS_STRIDE)        // 33280
#define HS_BYTES  (B_ * HS_STRIDE)        // 66560
#define TI_BYTES  (8 * 16 * 16 * 4)       // 8192: one f32 tile per wave
#define SMEM_BYTES (XS_BYTES + HS_BYTES + TI_BYTES)

__launch_bounds__(256, 1)
__global__ void lstm_persist_kernel(const __bf16* __restrict__ xb,    // [B][S][D]
                                    const __bf16* __restrict__ wih,   // [4H][D]
                                    const __bf16* __restrict__ whh,   // [4H][H]
                                    const float*  __restrict__ bsum,  // [4H]
                                    __bf16* __restrict__ hbuf,        // [2][B][H]
                                    float*  __restrict__ c,           // [B][H]
                                    float*  __restrict__ out,         // [B][S][H]
                                    unsigned* __restrict__ bar) {
    __shared__ __attribute__((aligned(16))) unsigned char smem[SMEM_BYTES];
    unsigned char* xs  = smem;                       // staged x_t   [32][1040B]
    unsigned char* hs  = smem + XS_BYTES;            // staged h_t-1 [32][2080B]
    float* tiles = (float*)(smem + XS_BYTES + HS_BYTES);
    const unsigned smem_lds = (unsigned)(unsigned long long)(void*)&smem[0];

    const int wg   = blockIdx.x;           // 0..63 -> columns [wg*16, wg*16+16)
    const int tid  = threadIdx.x;
    const int lane = tid & 31;
    const int wave = tid >> 5;             // 0..7
    const int mt   = wave & 1;             // batch tile (rows 0-15 / 16-31)
    const int gate = wave >> 1;            // 0=i 1=f 2=g 3=o
    const int n    = lane & 15;            // output column within tile
    const int lh   = lane >> 4;            // lane half (A/B layout selector)
    const int m    = mt * 16 + n;          // batch row this lane loads for A
    const int wrow = gate * H_ + wg * 16 + n;   // W row == gate output column

    const __bf16* wih_r = wih + (size_t)wrow * D_;
    const __bf16* whh_r = whh + (size_t)wrow * H_;
    const float   bias  = bsum[wrow];
    const unsigned nwg  = gridDim.x;

    const unsigned char* xrowL = xs + m * XS_STRIDE;
    const unsigned char* hrowL = hs + m * HS_STRIDE;

    for (int t = 0; t < S_; ++t) {
        // --- TDM: stage x_t (32x512) and h_{t-1} (32x1024) into LDS ---
        if (tid < 32) {                    // wave 0 issues; TENSORcnt tracks
            tdm_load_2d(smem_lds + 0,
                        xb + (size_t)t * D_,          // row b at +b*S*D elems
                        D_, B_, D_, B_,
                        (unsigned long long)S_ * D_, true);
            tdm_load_2d(smem_lds + XS_BYTES,
                        hbuf + (size_t)(t & 1) * (B_ * H_),
                        H_, B_, H_, B_,
                        (unsigned long long)H_, true);
            __builtin_amdgcn_s_wait_tensorcnt(0);
        }
        __syncthreads();

        v8f acc;
        #pragma unroll
        for (int r = 0; r < 8; ++r) acc[r] = bias;  // C tile: every VGPR is col n

        // ---- x_t @ W_ih^T : K = 512. A from LDS, B (weights) from L2. ----
        #pragma unroll 4
        for (int k0 = 0; k0 < D_; k0 += 32) {
            int ka = k0 + lh * 8;          // A halves: K {ka..+7, ka+16..+23}
            v16bf a = load_frag(xrowL + ka * 2, xrowL + (ka + 16) * 2);
            v16bf b = load_frag(wih_r + ka, wih_r + ka + 16);
            acc = __builtin_amdgcn_wmma_f32_16x16x32_bf16(
                      false, a, false, b, (short)0, acc, false, false);
        }
        // ---- h_{t-1} @ W_hh^T : K = 1024 (h rows have a 16B pad at k=512) ----
        #pragma unroll 4
        for (int k0 = 0; k0 < H_; k0 += 32) {
            int ka = k0 + lh * 8;
            int kc = ka + 16;
            unsigned o0 = (unsigned)(ka * 2) + (ka >= 512 ? 16u : 0u);
            unsigned o1 = (unsigned)(kc * 2) + (kc >= 512 ? 16u : 0u);
            v16bf a = load_frag(hrowL + o0, hrowL + o1);
            v16bf b = load_frag(whh_r + ka, whh_r + kc);
            acc = __builtin_amdgcn_wmma_f32_16x16x32_bf16(
                      false, a, false, b, (short)0, acc, false, false);
        }

        // Stash this wave's gate tile: tiles[wave][row][col]; row = r + 8*lh
        float* tile = tiles + wave * 256;
        #pragma unroll
        for (int r = 0; r < 8; ++r) tile[(r + 8 * lh) * 16 + n] = acc[r];
        __syncthreads();

        // Elementwise LSTM cell for our 16 columns: 32 batch x 16 cols.
        __bf16* hwr = hbuf + (size_t)((t + 1) & 1) * (B_ * H_);
        #pragma unroll
        for (int e = tid; e < 512; e += 256) {
            int b_  = e >> 4;
            int col = e & 15;
            int mt2 = b_ >> 4;
            int idx = (b_ & 15) * 16 + col;
            float ig = tiles[(0 + mt2) * 256 + idx];
            float fg = tiles[(2 + mt2) * 256 + idx];
            float gg = tiles[(4 + mt2) * 256 + idx];
            float og = tiles[(6 + mt2) * 256 + idx];
            ig = 1.f / (1.f + __expf(-ig));
            fg = 1.f / (1.f + __expf(-fg));
            gg = tanhf(gg);
            og = 1.f / (1.f + __expf(-og));
            int    colg = wg * 16 + col;
            size_t ci   = (size_t)b_ * H_ + colg;
            float cn = fg * c[ci] + ig * gg;
            c[ci] = cn;
            float hn = og * tanhf(cn);
            hwr[ci] = (__bf16)hn;
            out[((size_t)b_ * S_ + t) * H_ + colg] = hn;
        }

        // Grid-wide release/acquire barrier before next timestep.
        __threadfence();
        __syncthreads();
        if (tid == 0) {
            unsigned target = nwg * (unsigned)(t + 1);
            __hip_atomic_fetch_add(bar, 1u, __ATOMIC_RELEASE,
                                   __HIP_MEMORY_SCOPE_AGENT);
            while (__hip_atomic_load(bar, __ATOMIC_ACQUIRE,
                                     __HIP_MEMORY_SCOPE_AGENT) < target) {
                __builtin_amdgcn_s_sleep(4);
            }
        }
        __syncthreads();
        __threadfence();
    }
}

extern "C" void kernel_launch(void* const* d_in, const int* in_sizes, int n_in,
                              void* d_out, int out_size, void* d_ws, size_t ws_size,
                              hipStream_t stream) {
    const float* x    = (const float*)d_in[0];   // [B,S,D]
    const float* h0   = (const float*)d_in[1];   // [B,H]
    const float* c0   = (const float*)d_in[2];   // [B,H]
    const float* W_ih = (const float*)d_in[3];   // [4H,D]
    const float* W_hh = (const float*)d_in[4];   // [4H,H]
    const float* b_ih = (const float*)d_in[5];   // [4H]
    const float* b_hh = (const float*)d_in[6];   // [4H]
    float* out = (float*)d_out;                  // [B,S,H]

    char* ws = (char*)d_ws;
    const size_t N_X   = (size_t)B_ * S_ * D_;
    const size_t N_WIH = (size_t)G4H * D_;
    const size_t N_WHH = (size_t)G4H * H_;
    __bf16*   xb    = (__bf16*)ws;                         ws += N_X   * 2;
    __bf16*   wihb  = (__bf16*)ws;                         ws += N_WIH * 2;
    __bf16*   whhb  = (__bf16*)ws;                         ws += N_WHH * 2;
    float*    bsum  = (float*)ws;                          ws += G4H * 4;
    __bf16*   hbuf  = (__bf16*)ws;                         ws += 2 * B_ * H_ * 2;
    float*    cbuf  = (float*)ws;                          ws += (size_t)B_ * H_ * 4;
    unsigned* bar   = (unsigned*)ws;

    cvt_f32_to_bf16<<<(int)((N_X   + 255) / 256), 256, 0, stream>>>(x,    xb,   (int)N_X);
    cvt_f32_to_bf16<<<(int)((N_WIH + 255) / 256), 256, 0, stream>>>(W_ih, wihb, (int)N_WIH);
    cvt_f32_to_bf16<<<(int)((N_WHH + 255) / 256), 256, 0, stream>>>(W_hh, whhb, (int)N_WHH);
    prep_kernel<<<(B_ * H_ + 255) / 256, 256, 0, stream>>>(
        b_ih, b_hh, bsum, h0, c0, hbuf, cbuf, bar);
    lstm_persist_kernel<<<H_ / 16, 256, 0, stream>>>(
        xb, wihb, whhb, bsum, hbuf, cbuf, out, bar);
}
